// DMTetGeometry_29987461660710
// MI455X (gfx1250) — matrix operand order repaired and checked by reading, not compiled, not run on hardware
//
#include <hip/hip_runtime.h>
#include <math.h>

typedef unsigned long long ull;
typedef __attribute__((ext_vector_type(16))) _Float16 v16h;
typedef __attribute__((ext_vector_type(8)))  float    v8f;

#define TILE 4096            // elements per block tile (256 thr * 16 items)
#define RTHREADS 256

__constant__ int c_tri[16][6] = {
    {-1,-1,-1,-1,-1,-1},{1,0,2,-1,-1,-1},{4,0,3,-1,-1,-1},{1,4,2,1,3,4},
    {3,1,5,-1,-1,-1},{2,3,0,2,5,3},{1,4,0,1,5,4},{4,2,5,-1,-1,-1},
    {4,5,2,-1,-1,-1},{4,1,0,4,5,1},{3,2,0,3,5,2},{1,3,5,-1,-1,-1},
    {4,1,2,4,3,1},{3,0,4,-1,-1,-1},{2,0,1,-1,-1,-1},{-1,-1,-1,-1,-1,-1}};
__constant__ int c_numtri[16] = {0,1,1,2,1,2,2,1,1,2,2,1,2,1,1,0};

// ---------------------------------------------------------------- key build
__global__ void build_keys(const int* __restrict__ tet, const float* __restrict__ sdf,
                           ull* __restrict__ keys, int Fn, int N) {
    int f = blockIdx.x * 256 + threadIdx.x;
    if (f >= Fn) return;
    int v0 = tet[f*4+0], v1 = tet[f*4+1], v2 = tet[f*4+2], v3 = tet[f*4+3];
    int o = (sdf[v0] > 0.f) + (sdf[v1] > 0.f) + (sdf[v2] > 0.f) + (sdf[v3] > 0.f);
    bool valid = (o > 0) && (o < 4);
    ull FILL = (ull)N * (ull)N;
    int ea[6] = {v0, v0, v0, v1, v1, v2};
    int eb[6] = {v1, v2, v3, v2, v3, v3};
    #pragma unroll
    for (int e = 0; e < 6; ++e) {
        int a = ea[e] < eb[e] ? ea[e] : eb[e];
        int b = ea[e] < eb[e] ? eb[e] : ea[e];
        keys[(long long)f*6 + e] = valid ? ((ull)a * (ull)N + (ull)b) : FILL;
    }
}

// ------------------------------------------------------- radix sort (4-bit)
__global__ void radix_hist(const ull* __restrict__ src, unsigned* __restrict__ counts,
                           long long K, int shift, int numBlocks) {
    __shared__ unsigned h[16];
    if (threadIdx.x < 16) h[threadIdx.x] = 0;
    __syncthreads();
    long long base = (long long)blockIdx.x * TILE;
    for (int it = 0; it < 16; ++it) {
        long long i = base + it*256 + threadIdx.x;
        if (i + 4096 < K) __builtin_prefetch(&src[i + 4096], 0, 0);  // global_prefetch_b8
        if (i < K) {
            int d = (int)((src[i] >> shift) & 15ull);
            atomicAdd(&h[d], 1u);
        }
    }
    __syncthreads();
    if (threadIdx.x < 16) counts[(long long)threadIdx.x * numBlocks + blockIdx.x] = h[threadIdx.x];
}

// single-block exclusive scan (n up to ~32K)
__global__ void scan_single(const unsigned* __restrict__ in, unsigned* __restrict__ out, int n) {
    __shared__ unsigned tmp[1024];
    __shared__ unsigned carry;
    if (threadIdx.x == 0) carry = 0;
    __syncthreads();
    for (int base = 0; base < n; base += 1024) {
        int i = base + (int)threadIdx.x;
        unsigned v = (i < n) ? in[i] : 0u;
        tmp[threadIdx.x] = v;
        __syncthreads();
        for (int off = 1; off < 1024; off <<= 1) {
            unsigned a = (threadIdx.x >= (unsigned)off) ? tmp[threadIdx.x - off] : 0u;
            __syncthreads();
            tmp[threadIdx.x] += a;
            __syncthreads();
        }
        unsigned incl = tmp[threadIdx.x];
        unsigned c = carry;
        if (i < n) out[i] = c + incl - v;
        unsigned total = tmp[1023];
        __syncthreads();
        if (threadIdx.x == 0) carry = c + total;
        __syncthreads();
    }
}

// stable scatter: wave32 ballot ranking + cross-wave LDS combine
__global__ void radix_scatter(const ull* __restrict__ src, ull* __restrict__ dst,
                              const unsigned* __restrict__ scanned,
                              long long K, int shift, int numBlocks) {
    __shared__ unsigned waveCnt[8][16];
    __shared__ unsigned waveBase[8][16];
    __shared__ unsigned running[16];
    __shared__ unsigned iterTot[16];
    int tid = threadIdx.x, lane = tid & 31, wave = tid >> 5;
    if (tid < 16) running[tid] = 0;
    __syncthreads();
    long long base = (long long)blockIdx.x * TILE;
    unsigned laneMaskLt = (1u << lane) - 1u;
    for (int it = 0; it < 16; ++it) {
        long long i = base + it*256 + tid;
        bool active = i < K;
        ull key = active ? src[i] : ~0ull;
        int d = (int)((key >> shift) & 15ull);
        unsigned myRank = 0;
        #pragma unroll
        for (int dd = 0; dd < 16; ++dd) {
            unsigned b32 = (unsigned)__ballot(active && (d == dd));
            if (lane == 0) waveCnt[wave][dd] = __popc(b32);
            if (d == dd) myRank = __popc(b32 & laneMaskLt);
        }
        __syncthreads();
        if (tid < 16) {
            unsigned r = 0;
            for (int w = 0; w < 8; ++w) { waveBase[w][tid] = r; r += waveCnt[w][tid]; }
            iterTot[tid] = r;
        }
        __syncthreads();
        if (active) {
            unsigned rank = running[d] + waveBase[wave][d] + myRank;
            unsigned gbase = scanned[(long long)d * numBlocks + blockIdx.x];
            dst[(long long)gbase + rank] = key;
        }
        __syncthreads();
        if (tid < 16) running[tid] += iterTot[tid];
        __syncthreads();
    }
}

// ------------------------------------------------ unique (head flags + compaction)
__global__ void heads_scan1(const ull* __restrict__ s, unsigned* __restrict__ pref,
                            unsigned* __restrict__ bsums, long long K) {
    __shared__ unsigned ts[256];
    long long base = (long long)blockIdx.x * TILE;
    int tid = threadIdx.x;
    unsigned local[16];
    unsigned cnt = 0;
    #pragma unroll
    for (int j = 0; j < 16; ++j) {
        long long i = base + (long long)tid*16 + j;
        unsigned hf = 0;
        if (i < K) hf = (i == 0) || (s[i] != s[i-1]);
        local[j] = cnt;
        cnt += hf;
    }
    ts[tid] = cnt;
    __syncthreads();
    for (int off = 1; off < 256; off <<= 1) {
        unsigned a = (tid >= off) ? ts[tid-off] : 0u;
        __syncthreads();
        ts[tid] += a;
        __syncthreads();
    }
    unsigned excl = ts[tid] - cnt;
    #pragma unroll
    for (int j = 0; j < 16; ++j) {
        long long i = base + (long long)tid*16 + j;
        if (i < K) pref[i] = excl + local[j];
    }
    if (tid == 255) bsums[blockIdx.x] = ts[255];
}

__global__ void fill_uk(ull* __restrict__ uk, long long K, ull FILL) {
    long long i = (long long)blockIdx.x * 256 + threadIdx.x;
    if (i < K) uk[i] = FILL;
}

__global__ void scatter_uk(const ull* __restrict__ s, const unsigned* __restrict__ pref,
                           const unsigned* __restrict__ bbase, ull* __restrict__ uk, long long K) {
    long long i = (long long)blockIdx.x * 256 + threadIdx.x;
    if (i >= K) return;
    ull v = s[i];
    bool head = (i == 0) || (s[i-1] != v);
    if (head) {
        long long blk = i >> 12;
        uk[(long long)bbase[blk] + pref[i]] = v;
    }
}

// ------------------------------------------------ inverse via lower_bound in L2
__global__ void inv_kernel(const int* __restrict__ tet, const float* __restrict__ sdf,
                           const ull* __restrict__ uk, unsigned* __restrict__ inv,
                           int Fn, int N, long long K) {
    int f = blockIdx.x * 256 + threadIdx.x;
    if (f >= Fn) return;
    int v0 = tet[f*4+0], v1 = tet[f*4+1], v2 = tet[f*4+2], v3 = tet[f*4+3];
    int o = (sdf[v0] > 0.f) + (sdf[v1] > 0.f) + (sdf[v2] > 0.f) + (sdf[v3] > 0.f);
    bool valid = (o > 0) && (o < 4);
    ull FILL = (ull)N * (ull)N;
    int ea[6] = {v0, v0, v0, v1, v1, v2};
    int eb[6] = {v1, v2, v3, v2, v3, v3};
    #pragma unroll
    for (int e = 0; e < 6; ++e) {
        int a = ea[e] < eb[e] ? ea[e] : eb[e];
        int b = ea[e] < eb[e] ? eb[e] : ea[e];
        ull key = valid ? ((ull)a * (ull)N + (ull)b) : FILL;
        long long lo = 0, hi = K;
        while (lo < hi) {                       // lower_bound: uk resident in 192MB L2
            long long mid = (lo + hi) >> 1;
            if (uk[mid] < key) lo = mid + 1; else hi = mid;
        }
        inv[(long long)f*6 + e] = (unsigned)lo;
    }
}

// ------------------------------------------------ WMMA tensor-core mask scan
// 256-element inclusive prefix per 16x16 chunk:  P = X*U + Lstrict*(X*J)
__global__ void mask_scan_wmma(const ull* __restrict__ uk, const float* __restrict__ sdf,
                               unsigned* __restrict__ incl, unsigned* __restrict__ bsums,
                               long long K, int N) {
    __shared__ _Float16 sm[TILE];
    __shared__ float    sp[TILE];
    __shared__ float chunkSums[16];
    __shared__ float chunkBase[16];
    const ull FILL = (ull)N * (ull)N;
    int tid = threadIdx.x;
    long long base = (long long)blockIdx.x * TILE;

    for (int j = tid; j < TILE; j += 256) {
        long long i = base + j;
        float mval = 0.f;
        if (i < K) {
            ull k = uk[i];
            if (k < FILL) {
                ull a = k / (ull)N, b = k % (ull)N;
                if (a > (ull)(N-1)) a = N-1;
                if (b > (ull)(N-1)) b = N-1;
                int oa = sdf[a] > 0.f, ob = sdf[b] > 0.f;
                mval = (oa + ob == 1) ? 1.f : 0.f;
            }
        }
        sm[j] = (_Float16)mval;
    }
    __syncthreads();

    int wave = tid >> 5, lane = tid & 31;
    int lrow = lane & 15;
    bool hi = lane >= 16;

    // constant fragments (B-layout: half h -> K = h + (hi?16:0), N = lrow)
    v16h Ub = {}, Jb = {};
    #pragma unroll
    for (int h = 0; h < 16; ++h) {
        int k = h + (hi ? 16 : 0);
        Jb[h] = (k < 16) ? (_Float16)1.f : (_Float16)0.f;
        Ub[h] = (k < 16 && k <= lrow) ? (_Float16)1.f : (_Float16)0.f;
    }
    // A-layout: half h -> K = h + (hi?16:0), M = lrow
    v16h Ls = {};
    #pragma unroll
    for (int h = 0; h < 16; ++h) {
        int k = h + (hi ? 16 : 0);
        Ls[h] = (k < lrow) ? (_Float16)1.f : (_Float16)0.f;  // strict lower triangle
    }

    for (int c = wave*2; c < wave*2 + 2; ++c) {
        int cb = c * 256;
        v16h Xa = {};
        #pragma unroll
        for (int h = 0; h < 16; ++h) {
            int k = h + (hi ? 16 : 0);
            Xa[h] = (k < 16) ? sm[cb + lrow*16 + k] : (_Float16)0.f;
        }
        v8f zero = {};
        v8f Y  = __builtin_amdgcn_wmma_f32_16x16x32_f16(false, Xa, false, Jb, (short)0, zero, false, false);
        v8f C1 = __builtin_amdgcn_wmma_f32_16x16x32_f16(false, Xa, false, Ub, (short)0, zero, false, false);
        // D-layout(Y) -> B-layout: lanes<16 need rows 0..15 of own column
        float part[8];
        #pragma unroll
        for (int v = 0; v < 8; ++v) part[v] = __shfl(Y[v], lane ^ 16, 32);
        v16h Yb = {};
        if (!hi) {
            #pragma unroll
            for (int h = 0; h < 8; ++h)  Yb[h] = (_Float16)Y[h];
            #pragma unroll
            for (int h = 8; h < 16; ++h) Yb[h] = (_Float16)part[h-8];
        }
        v8f P = __builtin_amdgcn_wmma_f32_16x16x32_f16(false, Ls, false, Yb, (short)0, C1, false, false);
        #pragma unroll
        for (int v = 0; v < 8; ++v) {
            int M = v + (hi ? 8 : 0);
            sp[cb + M*16 + lrow] = P[v];
        }
    }
    __syncthreads();
    if (tid < 16) chunkSums[tid] = sp[tid*256 + 255];
    __syncthreads();
    if (tid == 0) {
        float r = 0.f;
        for (int c = 0; c < 16; ++c) { chunkBase[c] = r; r += chunkSums[c]; }
    }
    __syncthreads();
    for (int j = tid; j < TILE; j += 256) {
        long long i = base + j;
        if (i < K) incl[i] = (unsigned)(sp[j] + chunkBase[j >> 8] + 0.5f);
    }
    if (tid == 0) bsums[blockIdx.x] = (unsigned)(chunkBase[15] + chunkSums[15] + 0.5f);
}

// mapping = mask ? cumsum-1 : -1 ; also emit mask_edges as float
__global__ void finalize_map(const ull* __restrict__ uk, const float* __restrict__ sdf,
                             unsigned* __restrict__ mapp, const unsigned* __restrict__ bbase,
                             float* __restrict__ maskOut, long long K, int N) {
    long long i = (long long)blockIdx.x * 256 + threadIdx.x;
    if (i >= K) return;
    const ull FILL = (ull)N * (ull)N;
    ull k = uk[i];
    bool mask = false;
    if (k < FILL) {
        ull a = k / (ull)N, b = k % (ull)N;
        if (a > (ull)(N-1)) a = N-1;
        if (b > (ull)(N-1)) b = N-1;
        int oa = sdf[a] > 0.f, ob = sdf[b] > 0.f;
        mask = (oa + ob == 1);
    }
    unsigned inclv = mapp[i];
    long long blk = i >> 12;
    int mval = mask ? (int)(inclv + bbase[blk]) - 1 : -1;
    ((int*)mapp)[i] = mval;
    maskOut[i] = mask ? 1.f : 0.f;
}

// ------------------------------------------------ outputs
__global__ void verts_kernel(const ull* __restrict__ uk, const float* __restrict__ sdf,
                             const float* __restrict__ pos, const int* __restrict__ m,
                             float* __restrict__ verts, long long K, int N) {
    long long u = (long long)blockIdx.x * 256 + threadIdx.x;
    if (u >= K) return;
    const ull FILL = (ull)N * (ull)N;
    ull k = uk[u];
    if (k >= FILL) return;
    ull a = k / (ull)N, b = k % (ull)N;
    if (a > (ull)(N-1)) a = N-1;
    if (b > (ull)(N-1)) b = N-1;
    float s0 = sdf[a], s1 = sdf[b];
    if (((s0 > 0.f) + (s1 > 0.f)) != 1) return;
    float denom = s0 - s1;
    float w0 = (-s1) / denom, w1 = s0 / denom;
    long long idx = m[u];
    #pragma unroll
    for (int j = 0; j < 3; ++j)
        verts[idx*3 + j] = pos[a*3 + j] * w0 + pos[b*3 + j] * w1;
}

__global__ void faces_kernel(const int* __restrict__ tet, const float* __restrict__ sdf,
                             const unsigned* __restrict__ inv, const int* __restrict__ m,
                             float* __restrict__ facesOut, float* __restrict__ fmaskOut, int Fn) {
    int f = blockIdx.x * 256 + threadIdx.x;
    if (f >= Fn) return;
    int o0 = sdf[tet[f*4+0]] > 0.f, o1 = sdf[tet[f*4+1]] > 0.f;
    int o2 = sdf[tet[f*4+2]] > 0.f, o3 = sdf[tet[f*4+3]] > 0.f;
    int sum = o0 + o1 + o2 + o3;
    bool valid = (sum > 0) && (sum < 4);
    int tetindex = o0 + 2*o1 + 4*o2 + 8*o3;
    int numtri = c_numtri[tetindex];
    int idxmap[6];
    #pragma unroll
    for (int e = 0; e < 6; ++e) idxmap[e] = m[inv[(long long)f*6 + e]];
    #pragma unroll
    for (int t = 0; t < 2; ++t) {
        bool fm = valid && (t < numtri);
        fmaskOut[(long long)f*2 + t] = fm ? 1.f : 0.f;
        #pragma unroll
        for (int j = 0; j < 3; ++j) {
            int tv = c_tri[tetindex][t*3 + j];
            int c = tv < 0 ? 0 : (tv > 5 ? 5 : tv);
            facesOut[((long long)f*2 + t)*3 + j] = fm ? (float)idxmap[c] : -1.f;
        }
    }
}

__global__ void uvs_kernel(float* __restrict__ out, int NUV) {
    long long total = (long long)NUV * NUV * 4;
    long long r = (long long)blockIdx.x * 256 + threadIdx.x;
    if (r >= total) return;
    long long cell = r >> 2;
    int q = (int)(r & 3);
    int a = (int)(cell / NUV), b = (int)(cell % NUV);
    float invn = 1.0f / (float)NUV;
    float x = b * invn, y = a * invn, p = 0.9f * invn;
    float u = (q == 1 || q == 2) ? x + p : x;
    float v = (q >= 2) ? y + p : y;
    out[r*2] = u;
    out[r*2 + 1] = v;
}

__global__ void uvidx_kernel(float* __restrict__ out, int Fn) {
    long long total = (long long)Fn * 6;
    long long r = (long long)blockIdx.x * 256 + threadIdx.x;
    if (r >= total) return;
    long long f = r / 6;
    int rem = (int)(r % 6);
    int t = rem / 3, j = rem % 3;
    long long val = (j == 0) ? f*4 : ((j == 1) ? f*4 + t + 1 : f*4 + t + 2);
    out[r] = (float)val;
}

// ---------------------------------------------------------------- launch
extern "C" void kernel_launch(void* const* d_in, const int* in_sizes, int n_in,
                              void* d_out, int out_size, void* d_ws, size_t ws_size,
                              hipStream_t stream) {
    const float* pos = (const float*)d_in[0];
    const float* sdf = (const float*)d_in[1];
    const int*   tet = (const int*)d_in[2];
    int N  = in_sizes[0] / 3;
    int Fn = in_sizes[2] / 4;
    long long K = 6ll * Fn;
    ull FILL = (ull)N * (ull)N;

    // NUV = ceil(sqrt((2F+1)//2))
    long long tgt = (2ll*Fn + 1) / 2;
    int NUV = (int)ceil(sqrt((double)tgt));
    while ((long long)NUV * NUV < tgt) ++NUV;
    while ((long long)(NUV-1) * (NUV-1) >= tgt) --NUV;

    // workspace carve
    char* w = (char*)d_ws;
    auto carve = [&](size_t bytes) { void* p = (void*)w; w += (bytes + 255) & ~(size_t)255; return p; };
    ull*      keysA = (ull*)carve((size_t)K * 8);
    ull*      keysB = (ull*)carve((size_t)K * 8);
    unsigned* invb  = (unsigned*)carve((size_t)K * 4);
    unsigned* mapb  = (unsigned*)carve((size_t)K * 4);
    int Bsort = (int)((K + TILE - 1) / TILE);
    unsigned* counts     = (unsigned*)carve((size_t)16 * Bsort * 4);
    unsigned* countsScan = (unsigned*)carve((size_t)16 * Bsort * 4);
    unsigned* bsums      = (unsigned*)carve((size_t)Bsort * 4);
    unsigned* bbase      = (unsigned*)carve((size_t)Bsort * 4);

    long long gridK = (K + 255) / 256;
    int gridF = (Fn + 255) / 256;

    // output layout (all float, concatenated in return order)
    float* out = (float*)d_out;
    float* o_verts = out;                                  // K*3
    float* o_mask  = o_verts + K*3;                        // K
    float* o_faces = o_mask + K;                           // F*6
    float* o_fmask = o_faces + (long long)Fn*6;            // F*2
    float* o_uvs   = o_fmask + (long long)Fn*2;            // NUV*NUV*8
    float* o_uvidx = o_uvs + (long long)NUV*NUV*8;         // F*6

    // 1) build keys
    build_keys<<<gridF, 256, 0, stream>>>(tet, sdf, keysA, Fn, N);

    // 2) 9-pass LSD radix sort on bits [0,36)
    for (int p = 0; p < 9; ++p) {
        int shift = 4 * p;
        const ull* src = (p & 1) ? keysB : keysA;
        ull*       dst = (p & 1) ? keysA : keysB;
        radix_hist   <<<Bsort, 256, 0, stream>>>(src, counts, K, shift, Bsort);
        scan_single  <<<1, 1024, 0, stream>>>(counts, countsScan, 16 * Bsort);
        radix_scatter<<<Bsort, 256, 0, stream>>>(src, dst, countsScan, K, shift, Bsort);
    }
    ull* sorted = keysB;
    ull* uk     = keysA;   // keysA free after pass 8 read

    // 3) unique: head flags -> positions -> compact (FILL padded)
    fill_uk    <<<(unsigned)gridK, 256, 0, stream>>>(uk, K, FILL);
    heads_scan1<<<Bsort, 256, 0, stream>>>(sorted, invb, bsums, K);
    scan_single<<<1, 1024, 0, stream>>>(bsums, bbase, Bsort);
    scatter_uk <<<(unsigned)gridK, 256, 0, stream>>>(sorted, invb, bbase, uk, K);

    // 4) inverse mapping via binary search (uk L2-resident)
    inv_kernel<<<gridF, 256, 0, stream>>>(tet, sdf, uk, invb, Fn, N, K);

    // 5) mask cumsum via WMMA tensor-core scan, then finalize mapping (+mask output)
    mask_scan_wmma<<<Bsort, 256, 0, stream>>>(uk, sdf, mapb, bsums, K, N);
    scan_single   <<<1, 1024, 0, stream>>>(bsums, bbase, Bsort);
    finalize_map  <<<(unsigned)gridK, 256, 0, stream>>>(uk, sdf, mapb, bbase, o_mask, K, N);

    // 6) outputs
    hipMemsetAsync(o_verts, 0, (size_t)K * 3 * sizeof(float), stream);
    verts_kernel<<<(unsigned)gridK, 256, 0, stream>>>(uk, sdf, pos, (const int*)mapb, o_verts, K, N);
    faces_kernel<<<gridF, 256, 0, stream>>>(tet, sdf, invb, (const int*)mapb, o_faces, o_fmask, Fn);
    long long uvTotal = (long long)NUV * NUV * 4;
    uvs_kernel  <<<(unsigned)((uvTotal + 255) / 256), 256, 0, stream>>>(o_uvs, NUV);
    uvidx_kernel<<<(unsigned)(((long long)Fn*6 + 255) / 256), 256, 0, stream>>>(o_uvidx, Fn);

    (void)n_in; (void)out_size; (void)ws_size;
}
